// GNNlayer_77154792505361
// MI455X (gfx1250) — matrix operand order repaired and checked by reading, not compiled, not run on hardware
//
#include <hip/hip_runtime.h>
#include <cstdint>
#include <cstddef>

// Problem constants (match reference)
#define GN 8192   // rows
#define GD 256    // feature dim (in == out)
#define JT 32     // column-chunk (K of the PV WMMA)
#define WAVES 8   // waves per block

typedef __bf16 v16bf __attribute__((ext_vector_type(16)));
typedef float  v8f   __attribute__((ext_vector_type(8)));

union BF16x16 { v16bf v; uint4 q[2]; };
union US8     { unsigned short s[8]; uint4 q; };

__device__ __forceinline__ unsigned short f2bf(float f) {
  unsigned int u = __float_as_uint(f);
  u += 0x7FFFu + ((u >> 16) & 1u);   // round-to-nearest-even
  return (unsigned short)(u >> 16);
}

// Padé(3,2) tanh: abs err ~7e-5 for |t|<=0.125 (cosine sims / 8), ~2% at |t|=1
// -- far below bf16 resolution; avoids the libm tanhf polynomial call.
__device__ __forceinline__ float fast_tanh(float t) {
  const float t2 = t * t;
  return t * (27.0f + t2) * __frcp_rn(27.0f + 9.0f * t2);
}

__device__ __forceinline__ v8f wmma_bf16(v16bf a, v16bf b, v8f c) {
  // D = A(16x32 bf16) x B(32x16 bf16) + C(16x16 f32)
  return __builtin_amdgcn_wmma_f32_16x16x32_bf16(false, a, false, b, (short)0, c,
                                                 false, false);
}

// ---------------------------------------------------------------------------
// prep: per-row inverse norm; emit u = bf16(x/||x||) and xb = bf16(x)
// one wave per row, lane handles 8 contiguous floats
// ---------------------------------------------------------------------------
__global__ __launch_bounds__(256) void prep_rows_kernel(
    const float* __restrict__ x, unsigned short* __restrict__ u,
    unsigned short* __restrict__ xb) {
  const int row  = blockIdx.x * 8 + (threadIdx.x >> 5);
  const int lane = threadIdx.x & 31;
  const float4* xr = (const float4*)(x + (size_t)row * GD);
  float4 v0 = xr[lane * 2 + 0];
  float4 v1 = xr[lane * 2 + 1];
  float ss = v0.x * v0.x + v0.y * v0.y + v0.z * v0.z + v0.w * v0.w +
             v1.x * v1.x + v1.y * v1.y + v1.z * v1.z + v1.w * v1.w;
  ss += __shfl_xor(ss, 1, 32);
  ss += __shfl_xor(ss, 2, 32);
  ss += __shfl_xor(ss, 4, 32);
  ss += __shfl_xor(ss, 8, 32);
  ss += __shfl_xor(ss, 16, 32);
  const float rn = rsqrtf(ss);
  const float f[8] = {v0.x, v0.y, v0.z, v0.w, v1.x, v1.y, v1.z, v1.w};
  US8 pu, px;
#pragma unroll
  for (int i = 0; i < 8; ++i) {
    px.s[i] = f2bf(f[i]);
    pu.s[i] = f2bf(f[i] * rn);
  }
  ((uint4*)xb)[(size_t)row * 32 + lane] = px.q;
  ((uint4*)u)[(size_t)row * 32 + lane]  = pu.q;
}

__global__ __launch_bounds__(256) void prep_w_kernel(
    const float* __restrict__ W, unsigned short* __restrict__ Wb) {
  const int i = blockIdx.x * 256 + threadIdx.x;  // grid covers 256*256
  Wb[i] = f2bf(W[i]);
}

// ---------------------------------------------------------------------------
// fused cosine-attention: h = rownorm(exp(8*tanh(b*UU^T/8)) * A_shape) @ x
// One wave per 16-row tile; flash-style streaming over 32-column chunks.
// ---------------------------------------------------------------------------
__global__ __launch_bounds__(256) void fused_attn_kernel(
    const unsigned short* __restrict__ u,   // bf16, row-normalized x
    const unsigned short* __restrict__ xb,  // bf16 x (values)
    const float* __restrict__ Ashape,       // fp32 mask, N x N (one-pass read)
    const float* __restrict__ bscale,       // scalar b
    unsigned short* __restrict__ h) {       // bf16 out, N x GD
  __shared__ __align__(16) unsigned short u_lds[JT * GD];        // 16 KB
  __shared__ __align__(16) unsigned short xT_lds[GD * JT];       // 16 KB, xT[n][k]
  __shared__ __align__(16) unsigned short p_lds[WAVES * 16 * JT];// 8 KB

  const int tid  = threadIdx.x;
  const int wave = tid >> 5;
  const int lane = tid & 31;
  const int l15  = lane & 15;
  const int hi   = lane >> 4;  // half-wave select
  const int mbase = blockIdx.x * (16 * WAVES) + wave * 16;
  const float bs8 = bscale[0] * 0.125f;  // fold the /8 into the scale

  // A operand: this wave's 16 query rows of u, all K=256, WMMA A layout.
  BF16x16 areg[8];
  {
    const uint4* ug = (const uint4*)u;
    const int row = mbase + l15;
#pragma unroll
    for (int s = 0; s < 8; ++s) {
      const int o1 = 32 * s + hi * 16;       // contiguous 16 halves per lane along K
      areg[s].q[0] = ug[((size_t)row * GD + o1) >> 3];
      areg[s].q[1] = ug[((size_t)row * GD + o1 + 8) >> 3];
    }
  }

  v8f acc[16];                       // 16x256 f32 accumulator (C layout per d-tile)
  const v8f vzero = {0.f, 0.f, 0.f, 0.f, 0.f, 0.f, 0.f, 0.f};
#pragma unroll
  for (int t = 0; t < 16; ++t) acc[t] = vzero;
  float rsum[8] = {0.f, 0.f, 0.f, 0.f, 0.f, 0.f, 0.f, 0.f};

  for (int jb = 0; jb < GN; jb += JT) {
    __syncthreads();  // previous iteration done reading LDS
    // Stage u chunk (contiguous) and x chunk transposed (for PV B operand).
    {
      const uint4* usrc = (const uint4*)(u + (size_t)jb * GD);
      uint4* udst = (uint4*)u_lds;
#pragma unroll
      for (int k = 0; k < 4; ++k) udst[tid + 256 * k] = usrc[tid + 256 * k];
      const uint4* xsrc = (const uint4*)(xb + (size_t)jb * GD);
#pragma unroll
      for (int k = 0; k < 4; ++k) {
        const int f  = tid + 256 * k;
        const int kk = f >> 5;       // row within chunk (0..31)
        const int c8 = f & 31;       // group of 8 columns
        uint4 val = xsrc[f];
        const unsigned short* hs = (const unsigned short*)&val;
#pragma unroll
        for (int i = 0; i < 8; ++i) xT_lds[(c8 * 8 + i) * JT + kk] = hs[i];
      }
    }
    if (jb + JT < GN) __builtin_prefetch(u + (size_t)(jb + JT) * GD, 0, 1);
    __syncthreads();

    // S = U_tile (16x256) @ u_chunk^T (256x32) -> two 16x16 tiles
    v8f s0 = vzero, s1 = vzero;
    const uint4* ul = (const uint4*)u_lds;
#pragma unroll
    for (int s = 0; s < 8; ++s) {
      BF16x16 b0, b1;
      const int off = 32 * s + hi * 16;  // contiguous 16 halves per lane along K
      b0.q[0] = ul[(l15 * GD + off) >> 3];
      b0.q[1] = ul[(l15 * GD + off + 8) >> 3];
      b1.q[0] = ul[((16 + l15) * GD + off) >> 3];
      b1.q[1] = ul[((16 + l15) * GD + off + 8) >> 3];
      s0 = wmma_bf16(areg[s].v, b0.v, s0);
      s1 = wmma_bf16(areg[s].v, b1.v, s1);
    }

    // Elementwise: p = exp(8*tanh(b*s/8)) * mask ; online row-sum; stage P.
    unsigned short* pw = p_lds + wave * 16 * JT;
#pragma unroll
    for (int tile = 0; tile < 2; ++tile) {
      const v8f sv = tile ? s1 : s0;
      const int col = jb + tile * 16 + l15;
#pragma unroll
      for (int r = 0; r < 8; ++r) {
        const int m = r + 8 * hi;                         // C layout row
        const float th = fast_tanh(sv[r] * bs8);
        const float mask = Ashape[(size_t)(mbase + m) * GN + col];
        const float p = __expf(8.0f * th) * mask;
        rsum[r] += p;
        pw[m * JT + tile * 16 + l15] = f2bf(p);
      }
    }

    // Re-read P in WMMA A layout (wave-private buffer; no block barrier needed).
    BF16x16 pa;
    {
      const uint4* pl = (const uint4*)pw;
      const int o1 = hi * 16;
      pa.q[0] = pl[(l15 * JT + o1) >> 3];
      pa.q[1] = pl[(l15 * JT + 8 + o1) >> 3];
    }

    // acc += P (16x32) @ X_chunk (32x256), 16 d-tiles
    const uint4* xl = (const uint4*)xT_lds;
#pragma unroll
    for (int t = 0; t < 16; ++t) {
      BF16x16 bx;
      const int n = t * 16 + l15;
      bx.q[0] = xl[(n * JT + hi * 16) >> 3];
      bx.q[1] = xl[(n * JT + hi * 16 + 8) >> 3];
      acc[t] = wmma_bf16(pa.v, bx.v, acc[t]);
    }
  }

  // Row sums: reduce across the 16 lanes of each half-wave.
#pragma unroll
  for (int r = 0; r < 8; ++r) {
    float v = rsum[r];
    v += __shfl_xor(v, 1, 32);
    v += __shfl_xor(v, 2, 32);
    v += __shfl_xor(v, 4, 32);
    v += __shfl_xor(v, 8, 32);
    rsum[r] = 1.0f / v;
  }
  // Normalize and emit h (bf16).
#pragma unroll
  for (int t = 0; t < 16; ++t) {
#pragma unroll
    for (int r = 0; r < 8; ++r) {
      const int m = mbase + r + 8 * hi;
      const int c = t * 16 + l15;
      h[(size_t)m * GD + c] = f2bf(acc[t][r] * rsum[r]);
    }
  }
}

// ---------------------------------------------------------------------------
// out = h @ W^T + bias   (8192x256x256, WMMA bf16, f32 out)
// ---------------------------------------------------------------------------
__global__ __launch_bounds__(256) void linear_kernel(
    const unsigned short* __restrict__ h, const unsigned short* __restrict__ Wb,
    const float* __restrict__ bias, float* __restrict__ out) {
  const int tid  = threadIdx.x;
  const int wave = tid >> 5;
  const int lane = tid & 31;
  const int l15  = lane & 15;
  const int hi   = lane >> 4;
  const int mbase = blockIdx.x * (16 * WAVES) + wave * 16;

  BF16x16 areg[8];
  {
    const uint4* hg = (const uint4*)h;
    const int row = mbase + l15;
#pragma unroll
    for (int s = 0; s < 8; ++s) {
      const int o1 = 32 * s + hi * 16;
      areg[s].q[0] = hg[((size_t)row * GD + o1) >> 3];
      areg[s].q[1] = hg[((size_t)row * GD + o1 + 8) >> 3];
    }
  }

  const v8f vzero = {0.f, 0.f, 0.f, 0.f, 0.f, 0.f, 0.f, 0.f};
  v8f oacc[16];
#pragma unroll
  for (int t = 0; t < 16; ++t) oacc[t] = vzero;

  const uint4* wg = (const uint4*)Wb;
#pragma unroll
  for (int t = 0; t < 16; ++t) {
    const int orow = t * 16 + l15;  // B(k,n) = W[o][k], contiguous along k
#pragma unroll
    for (int s = 0; s < 8; ++s) {
      BF16x16 bw;
      const int off = 32 * s + hi * 16;
      bw.q[0] = wg[((size_t)orow * GD + off) >> 3];
      bw.q[1] = wg[((size_t)orow * GD + off + 8) >> 3];
      oacc[t] = wmma_bf16(areg[s].v, bw.v, oacc[t]);
    }
  }

#pragma unroll
  for (int t = 0; t < 16; ++t) {
    const int c = t * 16 + l15;
    const float bv = bias[c];
#pragma unroll
    for (int r = 0; r < 8; ++r) {
      const int m = mbase + r + 8 * hi;
      out[(size_t)m * GD + c] = oacc[t][r] + bv;
    }
  }
}

// ---------------------------------------------------------------------------
extern "C" void kernel_launch(void* const* d_in, const int* in_sizes, int n_in,
                              void* d_out, int out_size, void* d_ws, size_t ws_size,
                              hipStream_t stream) {
  (void)in_sizes; (void)n_in; (void)out_size; (void)ws_size;
  const float* x      = (const float*)d_in[0];  // [8192,256]
  const float* Ashape = (const float*)d_in[1];  // [8192,8192]
  const float* W      = (const float*)d_in[2];  // [256,256]
  const float* bias   = (const float*)d_in[3];  // [256]
  const float* b      = (const float*)d_in[4];  // scalar

  // workspace layout (bf16 operands): u | xb | h | Wb  = 12 MB + 128 KB
  unsigned short* u  = (unsigned short*)d_ws;
  unsigned short* xb = u + (size_t)GN * GD;
  unsigned short* h  = xb + (size_t)GN * GD;
  unsigned short* Wb = h + (size_t)GN * GD;

  prep_rows_kernel<<<GN / 8, 256, 0, stream>>>(x, u, xb);
  prep_w_kernel<<<(GD * GD) / 256, 256, 0, stream>>>(W, Wb);
  fused_attn_kernel<<<GN / (16 * WAVES), 256, 0, stream>>>(u, xb, Ashape, b, h);
  linear_kernel<<<GN / (16 * WAVES), 256, 0, stream>>>(h, Wb, bias, (float*)d_out);
}
